// LSTM_47837345743422
// MI455X (gfx1250) — matrix operand order repaired and checked by reading, not compiled, not run on hardware
//
#include <hip/hip_runtime.h>

typedef __attribute__((ext_vector_type(16))) __bf16 v16bf;
typedef __attribute__((ext_vector_type(8)))  float  v8f;

#define B_   64
#define T_   512
#define V_   256
#define MEM_ 1024

union FragU { uint4 u[2]; v16bf v; };

__device__ __forceinline__ float sigmoidf_(float x) { return 1.0f / (1.0f + __expf(-x)); }

// ---------------------------------------------------------------- init
__global__ void lstm_init_kernel(const float* __restrict__ s0,
                                 const float* __restrict__ s1,
                                 __bf16* __restrict__ h0i,
                                 __bf16* __restrict__ h1i,
                                 int* __restrict__ counters) {
  int idx = blockIdx.x * blockDim.x + threadIdx.x;
  if (idx < 128) counters[idx] = 0;
  if (idx < B_ * MEM_) {
    int b = idx / MEM_, m = idx % MEM_;
    h0i[idx] = (__bf16)s0[(size_t)b * 2 * MEM_ + MEM_ + m];
    h1i[idx] = (__bf16)s1[(size_t)b * 2 * MEM_ + MEM_ + m];
  }
}

// ---------------------------------------------------------------- f32 -> bf16
__global__ void cvt_f32_bf16_kernel(const float* __restrict__ src,
                                    __bf16* __restrict__ dst, int n) {
  int i = blockIdx.x * blockDim.x + threadIdx.x;
  if (i < n) dst[i] = (__bf16)src[i];
}

// ---------------------------------------------------------------- weight pack
// Pack row-major f32 W[Ktot, Ntot] into per-fragment bf16 blocks matching the
// k = 16h + s order that the A loads also use: for (ntile, ktile, lane) the lane's 16 consecutive
// K-values (n = ntile*16 + lane%16, kbase = ktile*32 + (lane/16)*16) are
// stored contiguously (32 bytes).
__global__ void pack_w_kernel(const float* __restrict__ W,
                              uint4* __restrict__ dst,
                              int Ntot, int KT, int NTtiles) {
  int idx = blockIdx.x * 256 + threadIdx.x;
  int total = NTtiles * KT * 32;
  if (idx >= total) return;
  int lane = idx & 31;
  int unit = idx >> 5;           // = nt*KT + kt
  int kt = unit % KT;
  int nt = unit / KT;
  int n  = nt * 16 + (lane & 15);
  int kb = kt * 32 + (lane >> 4) * 16;
  union { __bf16 h[16]; uint4 u[2]; } p;
#pragma unroll
  for (int j = 0; j < 16; ++j)
    p.h[j] = (__bf16)W[(size_t)(kb + j) * Ntot + n];
  dst[(size_t)idx * 2 + 0] = p.u[0];
  dst[(size_t)idx * 2 + 1] = p.u[1];
}

// ---------------------------------------------------------------- grid barrier
__device__ __forceinline__ void grid_barrier_(int* counter, int target, int nwg) {
  __threadfence();
  __syncthreads();
  if (threadIdx.x == 0) {
    __hip_atomic_fetch_add(counter, 1, __ATOMIC_RELEASE, __HIP_MEMORY_SCOPE_AGENT);
    while (__hip_atomic_load(counter, __ATOMIC_ACQUIRE, __HIP_MEMORY_SCOPE_AGENT) <
           target * nwg)
      __builtin_amdgcn_s_sleep(1);
  }
  __syncthreads();
  __threadfence();
}

// ---------------------------------------------------------------- fused LSTM
// Persistent pipelined kernel: 128 WGs. WGs 0..63 run layer 0 (work BEFORE the
// round barrier, producing h0[t]); WGs 64..127 run layer 1 (work AFTER the
// barrier, consuming h0[t] and h1[t-1]). 512 barriers total instead of 1022,
// and the two layers overlap, so round time ~= max(layer0, layer1).
//
// Per WG: 512 threads = 16 waves. Wave w: mtile = w&3 (16 batch rows),
// gate-pair = (w>>2)&1, K-half = w>>3 (split-K). Partial gate sums staged in
// zbuf[khalf][gate][row][col]; pointwise phase sums the two K-halves.
template <int DIN, bool AFTER>
__device__ __forceinline__ void lstm_run(
    uint4* ldsW, float (&zbuf)[2][4][64][16],
    const __bf16* __restrict__ xseq,   // [B, T, DIN]
    const __bf16* __restrict__ hinit,  // [B, MEM]
    const float*  __restrict__ state,  // [B, 2*MEM] (c = first half)
    const uint4*  __restrict__ wpack,  // packed bf16 W [(DIN+MEM), 4096]
    const float*  __restrict__ bias,   // [4*MEM]
    __bf16* __restrict__ hseq,         // out [B, T, MEM]
    float*  __restrict__ fstate,       // out [B, 2*MEM]
    int* __restrict__ counter, int ct, int nwg) {
  constexpr int KT = (DIN + MEM_) / 32;  // 40 or 64 k-tiles
  constexpr int KH = KT / 2;             // split-K half

  const int tid = threadIdx.x;
  // Stage this WG's 4 gate weight tiles into LDS (layout mirrors packed global).
  for (int i = tid; i < 4 * KT * 64; i += 512) {
    int g   = i / (KT * 64);
    int rem = i - g * (KT * 64);
    ldsW[i] = wpack[(size_t)(g * 64 + ct) * (KT * 64) + rem];
  }
  __syncthreads();

  const int lane  = tid & 31;
  const int wv    = tid >> 5;            // 0..15
  const int mtile = wv & 3;              // batch-row tile
  const int gbase = ((wv >> 2) & 1) * 2; // gates {0,1} or {2,3}
  const int khalf = wv >> 3;             // split-K half 0/1
  const int ktbeg = khalf * KH;
  const int half  = lane >> 4;
  const int o     = half * 16;           // A-frag K sub-offset
  const int r     = mtile * 16 + (lane & 15);

  // Pointwise ownership: thread -> (batch row pb, 2 columns pc0, pc0+1)
  const int pb  = tid >> 3;
  const int pc0 = (tid & 7) * 2;
  float creg[2], bi[2], bj[2], bf[2], bo[2];
#pragma unroll
  for (int q = 0; q < 2; ++q) {
    int col = ct * 16 + pc0 + q;
    creg[q] = state[(size_t)pb * 2 * MEM_ + col];
    bi[q] = bias[col];
    bj[q] = bias[MEM_ + col];
    bf[q] = bias[2 * MEM_ + col];
    bo[q] = bias[3 * MEM_ + col];
  }

  for (int rnd = 0; rnd < T_; ++rnd) {
    if (AFTER) grid_barrier_(counter, rnd + 1, nwg);
    {
      const int t = rnd;
      const __bf16* xr = xseq + ((size_t)r * T_ + t) * DIN;
      const __bf16* hr = (t == 0) ? (hinit + (size_t)r * MEM_)
                                  : (hseq + ((size_t)r * T_ + (t - 1)) * MEM_);
      v8f acc0 = {}, acc1 = {};
#pragma unroll 4
      for (int kk = 0; kk < KH; ++kk) {
        const int kt = ktbeg + kk;
        const int k0 = kt * 32;
        const __bf16* src;
        int ko;
        if (k0 < DIN) { src = xr; ko = k0; }
        else          { src = hr; ko = k0 - DIN; }
        FragU a;
        a.u[0] = *(const uint4*)(src + ko + o);
        a.u[1] = *(const uint4*)(src + ko + o + 8);
        FragU b0f, b1f;
        const uint4* bp0 = &ldsW[(((gbase + 0) * KT + kt) * 32 + lane) * 2];
        const uint4* bp1 = &ldsW[(((gbase + 1) * KT + kt) * 32 + lane) * 2];
        b0f.u[0] = bp0[0]; b0f.u[1] = bp0[1];
        b1f.u[0] = bp1[0]; b1f.u[1] = bp1[1];
        acc0 = __builtin_amdgcn_wmma_f32_16x16x32_bf16(false, a.v, false, b0f.v,
                                                       (short)0, acc0, false, false);
        acc1 = __builtin_amdgcn_wmma_f32_16x16x32_bf16(false, a.v, false, b1f.v,
                                                       (short)0, acc1, false, false);
      }
#pragma unroll
      for (int v = 0; v < 8; ++v) {
        zbuf[khalf][gbase + 0][mtile * 16 + v + 8 * half][lane & 15] = acc0[v];
        zbuf[khalf][gbase + 1][mtile * 16 + v + 8 * half][lane & 15] = acc1[v];
      }
      __syncthreads();

      // Pointwise LSTM cell update for our 2 (pb, col) elements.
#pragma unroll
      for (int q = 0; q < 2; ++q) {
        int col = pc0 + q;
        float iv = zbuf[0][0][pb][col] + zbuf[1][0][pb][col] + bi[q];
        float jv = zbuf[0][1][pb][col] + zbuf[1][1][pb][col] + bj[q];
        float fv = zbuf[0][2][pb][col] + zbuf[1][2][pb][col] + bf[q];
        float ov = zbuf[0][3][pb][col] + zbuf[1][3][pb][col] + bo[q];
        float c = creg[q] * sigmoidf_(fv + 1.0f) + sigmoidf_(iv) * tanhf(jv);
        float h = tanhf(c) * sigmoidf_(ov);
        creg[q] = c;
        hseq[((size_t)pb * T_ + t) * MEM_ + ct * 16 + col] = (__bf16)h;
        if (t == T_ - 1) {
          fstate[(size_t)pb * 2 * MEM_ + ct * 16 + col] = c;
          fstate[(size_t)pb * 2 * MEM_ + MEM_ + ct * 16 + col] = h;
        }
      }
    }
    if (!AFTER) grid_barrier_(counter, rnd + 1, nwg);
  }
}

__global__ __launch_bounds__(512, 1)
void lstm_fused_kernel(const __bf16* __restrict__ xbf,
                       const __bf16* __restrict__ h0i,
                       const __bf16* __restrict__ h1i,
                       const float*  __restrict__ state0,
                       const float*  __restrict__ state1,
                       const uint4*  __restrict__ w0p,
                       const uint4*  __restrict__ w1p,
                       const float*  __restrict__ b0,
                       const float*  __restrict__ b1,
                       __bf16* __restrict__ h0seq,
                       __bf16* __restrict__ h1seq,
                       float*  __restrict__ fs0,
                       float*  __restrict__ fs1,
                       int* __restrict__ counter) {
  __shared__ uint4 ldsW[4 * 64 * 64];            // 256KB (layer0 uses 160KB)
  __shared__ float zbuf[2][4][64][16];           // 32KB split-K gate staging
  const int layer = blockIdx.x >> 6;
  const int ct    = blockIdx.x & 63;             // mem-column tile
  const int nwg   = (int)gridDim.x;
  if (layer == 0) {
    lstm_run<V_, false>(ldsW, zbuf, xbf, h0i, state0, w0p, b0,
                        h0seq, fs0, counter, ct, nwg);
  } else {
    lstm_run<MEM_, true>(ldsW, zbuf, h0seq, h1i, state1, w1p, b1,
                         h1seq, fs1, counter, ct, nwg);
  }
}

// ---------------------------------------------------------------- projection
// out[B*T, 256] = h1[B*T, 1024] @ W_out + b_out ; one 16x16 tile per wave.
__global__ __launch_bounds__(256)
void proj_kernel(const __bf16* __restrict__ h1,   // [B*T, MEM]
                 const uint4* __restrict__ wp,    // packed bf16 W_out [MEM,256]
                 const float* __restrict__ bias,  // [256]
                 float* __restrict__ out) {       // [B*T, 256]
  constexpr int KT = MEM_ / 32;                   // 32
  const int tid  = threadIdx.x;
  const int lane = tid & 31;
  const int w    = blockIdx.x * 8 + (tid >> 5);   // global wave id
  const int mt   = w >> 4;                        // row tile 0..2047
  const int nt   = w & 15;                        // col tile 0..15
  const int half = lane >> 4;
  const int o    = half * 16;
  const int r    = mt * 16 + (lane & 15);
  const __bf16* ar = h1 + (size_t)r * MEM_;
  v8f acc = {};
#pragma unroll 4
  for (int kt = 0; kt < KT; ++kt) {
    FragU a, b;
    a.u[0] = *(const uint4*)(ar + kt * 32 + o);
    a.u[1] = *(const uint4*)(ar + kt * 32 + o + 8);
    const uint4* bp = wp + ((size_t)(nt * KT + kt) * 32 + lane) * 2;
    b.u[0] = bp[0]; b.u[1] = bp[1];
    acc = __builtin_amdgcn_wmma_f32_16x16x32_bf16(false, a.v, false, b.v,
                                                  (short)0, acc, false, false);
  }
  const int n = nt * 16 + (lane & 15);
  const float bv = bias[n];
#pragma unroll
  for (int v = 0; v < 8; ++v) {
    int row = mt * 16 + v + 8 * half;
    out[(size_t)row * V_ + n] = acc[v] + bv;
  }
}

// ---------------------------------------------------------------- launch
extern "C" void kernel_launch(void* const* d_in, const int* in_sizes, int n_in,
                              void* d_out, int out_size, void* d_ws, size_t ws_size,
                              hipStream_t stream) {
  const float* x      = (const float*)d_in[0];   // [64,512,256]
  const float* state0 = (const float*)d_in[1];   // [64,2048]
  const float* state1 = (const float*)d_in[2];   // [64,2048]
  const float* W0     = (const float*)d_in[3];   // [1280,4096]
  const float* b0     = (const float*)d_in[4];   // [4096]
  const float* W1     = (const float*)d_in[5];   // [2048,4096]
  const float* b1     = (const float*)d_in[6];   // [4096]
  const float* W_out  = (const float*)d_in[7];   // [1024,256]
  const float* b_out  = (const float*)d_in[8];   // [256]
  float* out = (float*)d_out;

  const int KT0 = (V_ + MEM_) / 32;      // 40
  const int KT1 = (MEM_ + MEM_) / 32;    // 64
  const int KTO = MEM_ / 32;             // 32

  // Carve workspace (256B aligned regions).
  char* ws = (char*)d_ws;
  size_t cur = 0;
  auto carve = [&](size_t bytes) {
    char* p = ws + cur;
    cur = (cur + bytes + 255) & ~(size_t)255;
    return p;
  };
  __bf16* xbf    = (__bf16*)carve((size_t)B_ * T_ * V_ * 2);        // 16.8 MB
  uint4*  w0p    = (uint4*) carve((size_t)256 * KT0 * 64 * 16);     // 10.5 MB
  uint4*  w1p    = (uint4*) carve((size_t)256 * KT1 * 64 * 16);     // 16.8 MB
  uint4*  wop    = (uint4*) carve((size_t)16  * KTO * 64 * 16);     // 0.5 MB
  __bf16* h0seq  = (__bf16*)carve((size_t)B_ * T_ * MEM_ * 2);      // 67 MB
  __bf16* h1seq  = (__bf16*)carve((size_t)B_ * T_ * MEM_ * 2);      // 67 MB
  __bf16* h0i    = (__bf16*)carve((size_t)B_ * MEM_ * 2);
  __bf16* h1i    = (__bf16*)carve((size_t)B_ * MEM_ * 2);
  int*    cnts   = (int*)   carve(512);
  (void)ws_size; (void)in_sizes; (void)n_in; (void)out_size;

  float* fs0 = out + (size_t)B_ * T_ * V_;                 // final_state layer0
  float* fs1 = fs0 + (size_t)B_ * 2 * MEM_;                // final_state layer1

  lstm_init_kernel<<<256, 256, 0, stream>>>(state0, state1, h0i, h1i, cnts);

  int nx = B_ * T_ * V_;
  cvt_f32_bf16_kernel<<<(nx + 255) / 256, 256, 0, stream>>>(x, xbf, nx);

  pack_w_kernel<<<(256 * KT0 * 32 + 255) / 256, 256, 0, stream>>>(W0, w0p, 4096, KT0, 256);
  pack_w_kernel<<<(256 * KT1 * 32 + 255) / 256, 256, 0, stream>>>(W1, w1p, 4096, KT1, 256);
  pack_w_kernel<<<(16 * KTO * 32 + 255) / 256, 256, 0, stream>>>(W_out, wop, 256, KTO, 16);

  // 128 WGs: 64 for layer0 (pre-barrier), 64 for layer1 (post-barrier).
  lstm_fused_kernel<<<128, 512, 0, stream>>>(xbf, h0i, h1i, state0, state1,
                                             w0p, w1p, b0, b1,
                                             h0seq, h1seq, fs0, fs1, cnts);

  // 2048 row tiles * 16 col tiles = 32768 tiles, 8 waves/WG -> 4096 WGs.
  proj_kernel<<<4096, 256, 0, stream>>>(h1seq, wop, b_out, out);
}
